// GlobalAttention_40243843563745
// MI455X (gfx1250) — matrix operand order repaired and checked by reading, not compiled, not run on hardware
//
#include <hip/hip_runtime.h>
#include <hip/hip_bf16.h>
#include <math.h>

// ---------------- CDNA5 WMMA / TDM types ----------------
typedef __attribute__((ext_vector_type(16))) __bf16 v16bf;
typedef __attribute__((ext_vector_type(8)))  float  v8f;
typedef unsigned int __attribute__((ext_vector_type(4))) v4u;
typedef int          __attribute__((ext_vector_type(4))) v4i;
typedef int          __attribute__((ext_vector_type(8))) v8i;

union BfFrag { v16bf v; uint4 q[2]; };

// Tensor Data Mover availability (device pass only).
#if defined(__HIP_DEVICE_COMPILE__) && __has_builtin(__builtin_amdgcn_tensor_load_to_lds) && __has_builtin(__builtin_amdgcn_s_wait_tensorcnt)
#define USE_TDM 1
#if __has_include(<hip/amd_detail/amd_gfx1250_TDM.h>)
#define TDM_SIX_ARGS 1   // amdgpu-toolchain (therock) 6-arg builtin
#endif
#endif

#define TMB 256          // block tile rows
#define TNB 128          // block tile cols
#define TK  32           // K step (one wmma_16x16x32 per subtile)
#define LDSK 40          // padded LDS row: 40 bf16 = 80 B (TDM pads 16B per 64B row)

#if USE_TDM
// Issue one TENSOR_LOAD_TO_LDS moving a 2-D bf16 tile (tile_d0 x tile_d1,
// row stride `stride0_elems` in memory) into LDS at byte offset `lds_off`,
// with LDS padding 16B after every 64B row (=> 80B pitch == LDSK*2).
// D# layout per CDNA5 ISA ch.8 (group0 128b, group1 256b; groups 2/3 unused).
__device__ __forceinline__ void tdm_load_tile_2d(unsigned lds_off, const __bf16* gptr,
                                                 unsigned tile_d0, unsigned tile_d1,
                                                 unsigned long long stride0_elems)
{
    unsigned long long ga = (unsigned long long)(uintptr_t)gptr;
    v4u g0;
    g0[0] = 1u;                                            // count=1, user D#
    g0[1] = lds_off;                                       // lds_addr (bytes)
    g0[2] = (unsigned)(ga & 0xFFFFFFFFull);                // global_addr[31:0]
    g0[3] = (unsigned)((ga >> 32) & 0x1FFFFFFull)          // global_addr[56:32]
          | (2u << 30);                                    // type = 2 ("image")
    v8i g1;
    g1[0] = (int)((1u << 16)       // data_size = 2 bytes
                | (1u << 20)       // pad_enable
                | (3u << 22)       // pad_interval: 16 DWORDs (one 64B row)
                | (3u << 25));     // pad_amount: 4 DWORDs (16B)
    // tensor dims == tile dims (tile fully in-bounds; no OOB clipping needed)
    g1[1] = (int)((tile_d0 & 0xFFFFu) << 16);              // tensor_dim0[15:0]
    g1[2] = (int)((tile_d0 >> 16) | ((tile_d1 & 0xFFFFu) << 16)); // dim0 hi | dim1 lo
    g1[3] = (int)((tile_d1 >> 16) | (tile_d0 << 16));      // dim1 hi | tile_dim0
    g1[4] = (int)tile_d1;                                  // tile_dim1 (tile_dim2=0)
    g1[5] = (int)(unsigned)(stride0_elems & 0xFFFFFFFFull);// tensor_dim0_stride lo
    g1[6] = (int)(unsigned)((stride0_elems >> 32) & 0xFFFFull); // stride hi (dim1_stride=0)
    g1[7] = 0;
    v4i z4 = {0, 0, 0, 0};
#if TDM_SIX_ARGS
    v8i z8 = {0, 0, 0, 0, 0, 0, 0, 0};
    __builtin_amdgcn_tensor_load_to_lds(g0, g1, z4, z4, z8, 0);
#else
    __builtin_amdgcn_tensor_load_to_lds(g0, g1, z4, z4, 0);
#endif
}
#endif // USE_TDM

// =====================================================================
// Tiled bf16 GEMM, fp32 accumulate, v_wmma_f32_16x16x32_bf16.
//   C[m][n] = sum_k A[m][k] * Bc[n][k]   (Bc = B stored column-major [N][K])
// Block 256x128, 8 waves (4Mx2N), wave tile 64x64 => 16 WMMA / wave / K-step.
// Double-buffered LDS; staging via TDM (TENSORcnt) or manual fallback.
// Epilogue: optional bf16 residual add, bf16 store, fp32 gamma*val + x store.
// All dims assumed multiples of tile sizes (true for this problem).
// =====================================================================
__global__ __launch_bounds__(256)
void gemm_bf16_wmma(const __bf16* __restrict__ A, const __bf16* __restrict__ Bc,
                    int K, int lda, int ldb, int ldc,
                    long long strideA, long long strideB, long long strideC,
                    const __bf16* __restrict__ residBf,
                    __bf16* __restrict__ outBf,
                    float* __restrict__ outF,
                    const float* __restrict__ xResid,
                    const float* __restrict__ gammaPtr)
{
    __shared__ __bf16 smA[2][TMB][LDSK];   // 2*256*40*2 = 40960 B
    __shared__ __bf16 smB[2][TNB][LDSK];   // 2*128*40*2 = 20480 B

    const int b = blockIdx.z;
    A  += (long long)b * strideA;
    Bc += (long long)b * strideB;
    const long long coff = (long long)b * strideC;

    const int m0   = blockIdx.y * TMB;
    const int n0   = blockIdx.x * TNB;
    const int tid  = threadIdx.x;
    const int lane = tid & 31;
    const int wave = tid >> 5;
    const int wm   = wave >> 1;      // 0..3 : 64-row slab
    const int wn   = wave & 1;       // 0..1 : 64-col slab
    const int half = lane >> 4;
    const int l16  = lane & 15;

    const v8f vzero = {0.f,0.f,0.f,0.f,0.f,0.f,0.f,0.f};
    v8f acc[4][4];
    #pragma unroll
    for (int i = 0; i < 4; ++i)
        #pragma unroll
        for (int j = 0; j < 4; ++j) acc[i][j] = vzero;

    auto stage = [&](int kt, int s) {
#if USE_TDM
        if (wave == 0) {
            tdm_load_tile_2d((unsigned)(uintptr_t)(&smA[s][0][0]),
                             A + (long long)m0 * lda + (long long)kt * TK,
                             TK, TMB, (unsigned long long)lda);
            tdm_load_tile_2d((unsigned)(uintptr_t)(&smB[s][0][0]),
                             Bc + (long long)n0 * ldb + (long long)kt * TK,
                             TK, TNB, (unsigned long long)ldb);
        }
#else
        const int k0 = kt * TK;
        #pragma unroll
        for (int it = 0; it < (TMB * TK / 8) / 256; ++it) {   // 4 iters
            int idx = tid + it * 256;
            int r   = idx >> 2;
            int c8  = (idx & 3) * 8;
            *(uint4*)&smA[s][r][c8] = *(const uint4*)(A + (long long)(m0 + r) * lda + k0 + c8);
        }
        #pragma unroll
        for (int it = 0; it < (TNB * TK / 8) / 256; ++it) {   // 2 iters
            int idx = tid + it * 256;
            int r   = idx >> 2;
            int c8  = (idx & 3) * 8;
            *(uint4*)&smB[s][r][c8] = *(const uint4*)(Bc + (long long)(n0 + r) * ldb + k0 + c8);
        }
#endif
    };

    const int nk = K / TK;
    int sel = 0;
    stage(0, 0);

    for (int ki = 0; ki < nk; ++ki) {
        const bool hasNext = (ki + 1) < nk;
        if (hasNext) stage(ki + 1, sel ^ 1);   // prefetch next tiles (other buffer)
#if USE_TDM
        if (wave == 0) {
            // 2 TDM ops per stage; allow the just-issued pair to stay in flight.
            if (hasNext) __builtin_amdgcn_s_wait_tensorcnt(2);
            else         __builtin_amdgcn_s_wait_tensorcnt(0);
        }
#endif
        __syncthreads();

        // ---- fragments per ISA 16-bit A / B VGPR striping ----
        BfFrag fa[4], fb[4];
        #pragma unroll
        for (int sm_ = 0; sm_ < 4; ++sm_) {
            int m = wm * 64 + sm_ * 16 + l16;
            fa[sm_].q[0] = *(const uint4*)&smA[sel][m][half * 8];      // K: half*8 .. +7
            fa[sm_].q[1] = *(const uint4*)&smA[sel][m][16 + half * 8]; // K: 16+half*8 .. +7
        }
        #pragma unroll
        for (int sn_ = 0; sn_ < 4; ++sn_) {
            int n = wn * 64 + sn_ * 16 + l16;
            fb[sn_].q[0] = *(const uint4*)&smB[sel][n][half * 8];      // K: half*8 .. +7
            fb[sn_].q[1] = *(const uint4*)&smB[sel][n][16 + half * 8]; // K: 16+half*8 .. +7
        }

        // ---- 16 WMMAs ----
        #pragma unroll
        for (int sm_ = 0; sm_ < 4; ++sm_)
            #pragma unroll
            for (int sn_ = 0; sn_ < 4; ++sn_)
                acc[sm_][sn_] = __builtin_amdgcn_wmma_f32_16x16x32_bf16(
                    false, fa[sm_].v, false, fb[sn_].v,
                    (short)0, acc[sm_][sn_], false, false);

        __syncthreads();
        sel ^= 1;
    }

    // ---- epilogue: C/D layout => VGPR r holds (M = r + 8*half, N = l16) ----
    const float gma = gammaPtr ? gammaPtr[0] : 1.0f;
    #pragma unroll
    for (int sm_ = 0; sm_ < 4; ++sm_) {
        #pragma unroll
        for (int sn_ = 0; sn_ < 4; ++sn_) {
            int mgBase = m0 + wm * 64 + sm_ * 16 + half * 8;
            int ng     = n0 + wn * 64 + sn_ * 16 + l16;
            #pragma unroll
            for (int r = 0; r < 8; ++r) {
                long long cidx = coff + (long long)(mgBase + r) * ldc + ng;
                float v = acc[sm_][sn_][r];
                if (residBf) v += (float)residBf[cidx];
                if (outBf)   outBf[cidx] = (__bf16)v;
                if (outF)    outF[cidx]  = gma * v + (xResid ? xResid[cidx] : 0.0f);
            }
        }
    }
}

// =====================================================================
// Row softmax: one 256-thread block per row of a [rows][Ncols] bf16 matrix.
// =====================================================================
__global__ __launch_bounds__(256)
void softmax_rows_bf16(const __bf16* __restrict__ in, __bf16* __restrict__ out, int Ncols)
{
    const long long row = blockIdx.x;
    const __bf16* rp = in  + row * (long long)Ncols;
    __bf16*       op = out + row * (long long)Ncols;
    const int tid = threadIdx.x;
    const int PT  = Ncols / 256;       // 16 for N=4096
    float vals[16];

    float mx = -3.4e38f;
    for (int i = 0; i < PT; ++i) {
        float v = (float)rp[tid + i * 256];
        vals[i] = v;
        mx = fmaxf(mx, v);
    }
    __shared__ float red[256];
    red[tid] = mx; __syncthreads();
    for (int s = 128; s > 0; s >>= 1) {
        if (tid < s) red[tid] = fmaxf(red[tid], red[tid + s]);
        __syncthreads();
    }
    mx = red[0]; __syncthreads();

    float sum = 0.f;
    for (int i = 0; i < PT; ++i) { float e = __expf(vals[i] - mx); vals[i] = e; sum += e; }
    red[tid] = sum; __syncthreads();
    for (int s = 128; s > 0; s >>= 1) {
        if (tid < s) red[tid] += red[tid + s];
        __syncthreads();
    }
    const float inv = 1.0f / red[0];
    for (int i = 0; i < PT; ++i) op[tid + i * 256] = (__bf16)(vals[i] * inv);
}

// =====================================================================
// 1x1 conv projections: q^T [B][N][d], k^T [B][N][d], v [B][C][N] in bf16.
// =====================================================================
__global__ __launch_bounds__(256)
void qkv_kernel(const float* __restrict__ x,
                const float* __restrict__ Wq, const float* __restrict__ bq,
                const float* __restrict__ Wk, const float* __restrict__ bk,
                const float* __restrict__ Wv, const float* __restrict__ bv,
                __bf16* __restrict__ qT, __bf16* __restrict__ kT, __bf16* __restrict__ vout,
                int Bn, int C, int N, int Dq)
{
    long long t = (long long)blockIdx.x * blockDim.x + threadIdx.x;
    long long total = (long long)Bn * (2 * Dq + C) * N;
    if (t >= total) return;
    int n  = (int)(t % N);
    long long r = t / N;
    int oc = (int)(r % (2 * Dq + C));
    int b  = (int)(r / (2 * Dq + C));

    const float* xb = x + ((long long)b * C) * N;
    const float* w; float bias;
    if (oc < Dq)        { w = Wq + (long long)oc * C;            bias = bq[oc]; }
    else if (oc < 2*Dq) { w = Wk + (long long)(oc - Dq) * C;     bias = bk[oc - Dq]; }
    else                { w = Wv + (long long)(oc - 2*Dq) * C;   bias = bv[oc - 2*Dq]; }

    float acc = bias;
    #pragma unroll 4
    for (int c = 0; c < C; ++c) acc += w[c] * xb[(long long)c * N + n];

    __bf16 h = (__bf16)acc;
    if (oc < Dq)        qT[((long long)b * N + n) * Dq + oc] = h;
    else if (oc < 2*Dq) kT[((long long)b * N + n) * Dq + (oc - Dq)] = h;
    else                vout[((long long)b * C + (oc - 2*Dq)) * N + n] = h;
}

// =====================================================================
// bf16 square-matrix transpose (64x64 LDS tiles), per batch slice.
// =====================================================================
__global__ __launch_bounds__(256)
void transpose_bf16(const __bf16* __restrict__ in, __bf16* __restrict__ out,
                    int Ncols, long long bstride)
{
    __shared__ __bf16 sm[64][72];   // 144B row stride (16B multiple)
    in  += (long long)blockIdx.z * bstride;
    out += (long long)blockIdx.z * bstride;
    const int rb  = blockIdx.y * 64;
    const int cb  = blockIdx.x * 64;
    const int tid = threadIdx.x;

    #pragma unroll
    for (int it = 0; it < 2; ++it) {
        int idx = tid + it * 256;
        int rr  = idx >> 3;
        int c8  = (idx & 7) * 8;
        *(uint4*)&sm[rr][c8] = *(const uint4*)(in + (long long)(rb + rr) * Ncols + cb + c8);
    }
    __syncthreads();
    #pragma unroll
    for (int it = 0; it < 2; ++it) {
        int idx = tid + it * 256;
        int c   = idx >> 3;
        int r8  = (idx & 7) * 8;
        alignas(16) __bf16 tmp[8];
        #pragma unroll
        for (int j = 0; j < 8; ++j) tmp[j] = sm[r8 + j][c];
        *(uint4*)(out + (long long)(cb + c) * Ncols + rb + r8) = *(const uint4*)tmp;
    }
}

// =====================================================================
// Host-side orchestration
// =====================================================================
extern "C" void kernel_launch(void* const* d_in, const int* in_sizes, int n_in,
                              void* d_out, int out_size, void* d_ws, size_t ws_size,
                              hipStream_t stream)
{
    (void)in_sizes; (void)n_in; (void)out_size; (void)ws_size;
    const int Bn = 2, C = 256, H = 64, W = 64;
    const int N  = H * W;    // 4096
    const int Dq = C / 8;    // 32

    const float* x     = (const float*)d_in[0];
    const float* Wq    = (const float*)d_in[1];
    const float* bq    = (const float*)d_in[2];
    const float* Wk    = (const float*)d_in[3];
    const float* bk    = (const float*)d_in[4];
    const float* Wv    = (const float*)d_in[5];
    const float* bv    = (const float*)d_in[6];
    const float* gamma = (const float*)d_in[7];
    float* out = (float*)d_out;

    // workspace carve-up (bf16 buffers, 256B aligned)
    char* wsp = (char*)d_ws;
    auto takeBf = [&](size_t elems) {
        __bf16* p = (__bf16*)wsp;
        wsp += ((elems * 2 + 255) / 256) * 256;
        return p;
    };
    __bf16* qT   = takeBf((size_t)Bn * N * Dq);   //  0.5 MB
    __bf16* kT   = takeBf((size_t)Bn * N * Dq);   //  0.5 MB
    __bf16* vbf  = takeBf((size_t)Bn * C * N);    //  4   MB
    __bf16* Ebf  = takeBf((size_t)Bn * N * N);    // 64   MB  energy
    __bf16* EbfT = takeBf((size_t)Bn * N * N);    // 64   MB  energy^T, later attention
    __bf16* Xbf  = takeBf((size_t)Bn * N * N);    // 64   MB  e2, later energy'
    __bf16* Abf  = takeBf((size_t)Bn * N * N);    // 64   MB  softmax(e2)

    const long long sNN = (long long)N * N;
    const long long sND = (long long)N * Dq;
    const long long sCN = (long long)C * N;

    // 1) q/k/v projections
    {
        long long total = (long long)Bn * (2 * Dq + C) * N;
        int blocks = (int)((total + 255) / 256);
        qkv_kernel<<<blocks, 256, 0, stream>>>(x, Wq, bq, Wk, bk, Wv, bv,
                                               qT, kT, vbf, Bn, C, N, Dq);
    }
    // 2) energy = q^T k   (K = 32, single WMMA K-step)
    {
        dim3 g(N / TNB, N / TMB, Bn);
        gemm_bf16_wmma<<<g, 256, 0, stream>>>(qT, kT, Dq, Dq, Dq, N,
                                              sND, sND, sNN,
                                              nullptr, Ebf, nullptr, nullptr, nullptr);
    }
    // 3) E^T for the e2 GEMM's B operand
    {
        dim3 g(N / 64, N / 64, Bn);
        transpose_bf16<<<g, 256, 0, stream>>>(Ebf, EbfT, N, sNN);
    }
    // 4) e2 = E @ E   (B operand column-major = E^T row-major)
    {
        dim3 g(N / TNB, N / TMB, Bn);
        gemm_bf16_wmma<<<g, 256, 0, stream>>>(Ebf, EbfT, N, N, N, N,
                                              sNN, sNN, sNN,
                                              nullptr, Xbf, nullptr, nullptr, nullptr);
    }
    // 5) a = softmax(e2)
    softmax_rows_bf16<<<Bn * N, 256, 0, stream>>>(Xbf, Abf, N);
    // 6) energy' = E @ a^T + E   (a row-major [m][k] is exactly B column-major)
    {
        dim3 g(N / TNB, N / TMB, Bn);
        gemm_bf16_wmma<<<g, 256, 0, stream>>>(Ebf, Abf, N, N, N, N,
                                              sNN, sNN, sNN,
                                              Ebf, Xbf, nullptr, nullptr, nullptr);
    }
    // 7) attention = softmax(energy')  (reuse EbfT)
    softmax_rows_bf16<<<Bn * N, 256, 0, stream>>>(Xbf, EbfT, N);
    // 8) out = gamma * (v @ attention^T) + x   (fp32 store to d_out)
    {
        dim3 g(N / TNB, C / TMB, Bn);   // M = 256 -> one block row
        gemm_bf16_wmma<<<g, 256, 0, stream>>>(vbf, EbfT, N, N, N, N,
                                              sCN, sNN, sCN,
                                              nullptr, nullptr, out, x, gamma);
    }
}
